// BayesRNN_51719996178458
// MI455X (gfx1250) — matrix operand order repaired and checked by reading, not compiled, not run on hardware
//
#include <hip/hip_runtime.h>

#define BB 512
#define SS 2048
#define FF 64
#define HH 128
#define OO 32
#define BT 16     // batch rows per workgroup
#define NTHREADS 256

typedef __attribute__((ext_vector_type(16))) __bf16 v16bf;
typedef __attribute__((ext_vector_type(8)))  __bf16 v8bf;
typedef __attribute__((ext_vector_type(8)))  float  v8f;

// Branch-free tanh: native v_tanh_f32 if the builtin exists, else exp2+rcp.
__device__ __forceinline__ float fast_tanh(float v) {
#if __has_builtin(__builtin_amdgcn_tanhf)
    return __builtin_amdgcn_tanhf(v);
#else
    // tanh(v) = sign(v) * (1 - e) / (1 + e),  e = exp(-2|v|) = exp2(-2*log2e*|v|)
    const float a = __builtin_fabsf(v);
    const float e = __builtin_amdgcn_exp2f(-2.8853900817779268f * a);
    const float r = (1.0f - e) * __builtin_amdgcn_rcpf(1.0f + e);
    return __builtin_copysignf(r, v);
#endif
}

// A fragment (16x32 bf16): pA already points at (m*ld + half*8); fragment kc at
// element offset kc*32, with the second 8 elements at +16.
__device__ __forceinline__ v16bf load_a_frag(const __bf16* pA, int koff) {
    union { v16bf v; v8bf h[2]; } u;
    u.h[0] = *(const v8bf*)(pA + koff);
    u.h[1] = *(const v8bf*)(pA + koff + 16);
    return u.v;
}

// B fragment (32x16 bf16), B = W^T, W row-major [Nout][K], leading dim ldw.
__device__ __forceinline__ v16bf load_b_frag(const __bf16* W, int ldw, int nbase, int kbase, int lane) {
    const int n    = nbase + (lane & 15);
    const int half = lane >> 4;
    union { v16bf v; v8bf h[2]; } u;
    u.h[0] = *(const v8bf*)(W + n * ldw + kbase + half * 8);
    u.h[1] = *(const v8bf*)(W + n * ldw + kbase + half * 8 + 16);
    return u.v;
}

__global__ __launch_bounds__(NTHREADS) void bayes_rnn_fused(
    const float* __restrict__ x,    // [B,S,F]
    const float* __restrict__ Wih,  // [H,F]
    const float* __restrict__ bih,  // [H]
    const float* __restrict__ Whh,  // [H,H]
    const float* __restrict__ bhh,  // [H]
    const float* __restrict__ Who,  // [O,H]
    const float* __restrict__ bho,  // [O]
    float* __restrict__ out)        // [B,O]
{
    __shared__ __align__(32) __bf16 sWhh[HH * HH];      // 32 KB
    __shared__ __align__(32) __bf16 sWih[HH * FF];      // 16 KB
    __shared__ __align__(32) __bf16 sWho[OO * HH];      //  8 KB
    __shared__ __align__(32) __bf16 sX[2][BT * FF];     //  4 KB (double buffer)
    __shared__ __align__(32) __bf16 sH[2][BT * HH];     //  8 KB (double buffer)
    __shared__ float sBsum[HH];
    __shared__ float sBho[OO];

    const int tid  = threadIdx.x;
    const int wave = tid >> 5;
    const int lane = tid & 31;
    const int b0   = blockIdx.x * BT;

    // ---- one-time: weights -> bf16 LDS, biases, h0 = 0 ----
    for (int i = tid; i < HH * HH; i += NTHREADS) sWhh[i] = (__bf16)Whh[i];
    for (int i = tid; i < HH * FF; i += NTHREADS) sWih[i] = (__bf16)Wih[i];
    for (int i = tid; i < OO * HH; i += NTHREADS) sWho[i] = (__bf16)Who[i];
    if (tid < HH) sBsum[tid] = bih[tid] + bhh[tid];
    if (tid < OO) sBho[tid]  = bho[tid];
    for (int i = tid; i < BT * HH; i += NTHREADS) {
        sH[0][i] = (__bf16)0.0f;
        sH[1][i] = (__bf16)0.0f;
    }
    __syncthreads();

    // ---- per-wave resident B fragments (held in VGPRs for all 2048 steps) ----
    const int n0 = wave * 16;
    v16bf fWih[2], fWhh[4];
#pragma unroll
    for (int kc = 0; kc < 2; ++kc) fWih[kc] = load_b_frag(sWih, FF, n0, kc * 32, lane);
#pragma unroll
    for (int kc = 0; kc < 4; ++kc) fWhh[kc] = load_b_frag(sWhh, HH, n0, kc * 32, lane);
    const float bias = sBsum[n0 + (lane & 15)];

    // ---- precompute all per-thread pointers (hoisted out of the scan) ----
    const int m    = lane & 15;
    const int half = lane >> 4;
    const int n    = n0 + m;
    const int mb   = half * 8;

    // x loader: thread handles 4 consecutive fp32 of the 16x64 tile
    const int lidx = tid * 4;            // 0..1020
    const int lrow = lidx >> 6;
    const int lcol = lidx & 63;
    const float* xsrc = x + ((size_t)(b0 + lrow) * SS) * FF + lcol;  // advance by FF per step
    __bf16* xdst0 = &sX[0][lrow * FF + lcol];
    __bf16* xdst1 = &sX[1][lrow * FF + lcol];

    // A-fragment base pointers (per lane) for both buffers
    const __bf16* aX0 = &sX[0][m * FF + half * 8];
    const __bf16* aX1 = &sX[1][m * FF + half * 8];
    const __bf16* aH0 = &sH[0][m * HH + half * 8];
    const __bf16* aH1 = &sH[1][m * HH + half * 8];

    // h store base pointers (per lane): column n, rows mb..mb+7
    __bf16* hst0 = &sH[0][mb * HH + n];
    __bf16* hst1 = &sH[1][mb * HH + n];

    // ---- one timestep: read buffers (aX_r from xdst_w's buffer, aH_r), write hst_w ----
    auto step = [&](__bf16* xdst, const __bf16* aXr, const __bf16* aHr, __bf16* hstw) {
        // cooperative x_t tile load (b128), convert to bf16, prefetch next step
        const float4 v = *(const float4*)xsrc;
        xdst[0] = (__bf16)v.x; xdst[1] = (__bf16)v.y;
        xdst[2] = (__bf16)v.z; xdst[3] = (__bf16)v.w;
        __builtin_prefetch(xsrc + FF, 0, 3);
        xsrc += FF;
        __syncthreads();   // x_t visible; previous step's h writes visible

        // preload ALL A fragments first so ds_load latency overlaps the WMMA chain
        v16bf ax0 = load_a_frag(aXr, 0);
        v16bf ax1 = load_a_frag(aXr, 32);
        v16bf ah0 = load_a_frag(aHr, 0);
        v16bf ah1 = load_a_frag(aHr, 32);
        v16bf ah2 = load_a_frag(aHr, 64);
        v16bf ah3 = load_a_frag(aHr, 96);

        v8f acc;
#pragma unroll
        for (int i = 0; i < 8; ++i) acc[i] = bias;
        acc = __builtin_amdgcn_wmma_f32_16x16x32_bf16(false, ax0, false, fWih[0], (short)0, acc, false, false);
        acc = __builtin_amdgcn_wmma_f32_16x16x32_bf16(false, ax1, false, fWih[1], (short)0, acc, false, false);
        acc = __builtin_amdgcn_wmma_f32_16x16x32_bf16(false, ah0, false, fWhh[0], (short)0, acc, false, false);
        acc = __builtin_amdgcn_wmma_f32_16x16x32_bf16(false, ah1, false, fWhh[1], (short)0, acc, false, false);
        acc = __builtin_amdgcn_wmma_f32_16x16x32_bf16(false, ah2, false, fWhh[2], (short)0, acc, false, false);
        acc = __builtin_amdgcn_wmma_f32_16x16x32_bf16(false, ah3, false, fWhh[3], (short)0, acc, false, false);

        // tanh (native TRANS op) + bf16 store of h tile column
#pragma unroll
        for (int i = 0; i < 8; ++i) {
            hstw[i * HH] = (__bf16)fast_tanh(acc[i]);
        }
    };

    // ---- sequential scan: 2 steps per iteration removes the buffer toggle ----
    for (int t = 0; t < SS; t += 2) {
        step(xdst0, aX0, aH0, hst1);   // read buf0, write buf1
        step(xdst1, aX1, aH1, hst0);   // read buf1, write buf0
    }
    __syncthreads();
    // after an even number of steps the final h is in buffer 0

    // ---- output head: tanh(h_last @ Who^T + bho), O=32 -> waves 0,1 ----
    if (wave < 2) {
        const int no = wave * 16;
        const float ob = sBho[no + m];
        const __bf16* aHf = &sH[0][m * HH + half * 8];
        v8f acc;
#pragma unroll
        for (int i = 0; i < 8; ++i) acc[i] = ob;
#pragma unroll
        for (int kc = 0; kc < 4; ++kc) {
            v16bf a = load_a_frag(aHf, kc * 32);
            v16bf b = load_b_frag(sWho, HH, no, kc * 32, lane);
            acc = __builtin_amdgcn_wmma_f32_16x16x32_bf16(false, a, false, b, (short)0, acc, false, false);
        }
        const int nout = no + m;
#pragma unroll
        for (int i = 0; i < 8; ++i) {
            out[(size_t)(b0 + mb + i) * OO + nout] = fast_tanh(acc[i]);
        }
    }
}

extern "C" void kernel_launch(void* const* d_in, const int* in_sizes, int n_in,
                              void* d_out, int out_size, void* d_ws, size_t ws_size,
                              hipStream_t stream) {
    (void)in_sizes; (void)n_in; (void)d_ws; (void)ws_size; (void)out_size;
    const float* x    = (const float*)d_in[0];
    const float* Wih  = (const float*)d_in[1];
    const float* bih  = (const float*)d_in[2];
    const float* Whh  = (const float*)d_in[3];
    const float* bhh  = (const float*)d_in[4];
    const float* Who  = (const float*)d_in[5];
    const float* bho  = (const float*)d_in[6];
    float* out = (float*)d_out;

    dim3 grid(BB / BT);      // 32 workgroups
    dim3 block(NTHREADS);    // 8 wave32 waves
    bayes_rnn_fused<<<grid, block, 0, stream>>>(x, Wih, bih, Whh, bhh, Who, bho, out);
}
